// HalutMatmul_53747220742286
// MI455X (gfx1250) — matrix operand order repaired and checked by reading, not compiled, not run on hardware
//
#include <hip/hip_runtime.h>
#include <hip/hip_bf16.h>
#include <stdint.h>

typedef _Float16 v16h __attribute__((ext_vector_type(16)));
typedef float    v8f  __attribute__((ext_vector_type(8)));

#define N_      32768
#define D_      512
#define C_      64
#define K_      16
#define M_      128
#define NODES_  15
#define CK_     (C_ * K_)        // 1024 contraction dim
#define NCHUNK_ (CK_ / 32)       // 32 k-chunks of 32
#define MT_     (M_ / 16)        // 8 m-tiles
#define BFRAG_HALVES (MT_ * NCHUNK_ * 32 * 16)   // 131072 halves = 256KB

// ---------------------------------------------------------------------------
// Kernel 1: decision-tree argmax -> ksel[n*64+c] in [0,16)
// Matches reference exactly: h = I[:,dims]@S.T - T ; s = sign(h);
// b[k] = sum_l (bit ? s[node] : -s[node]) ; argmax(b / t) first-max tie-break.
// ---------------------------------------------------------------------------
__global__ __launch_bounds__(256) void halut_ksel(
    const float* __restrict__ I, const float* __restrict__ T,
    const int* __restrict__ dims, const float* __restrict__ temp,
    unsigned char* __restrict__ ksel)
{
  __shared__ float Ts[C_ * NODES_];   // 960 floats
  __shared__ int   Ds[C_ * 4];        // 256 ints
  int tid = threadIdx.x;
  for (int i = tid; i < C_ * NODES_; i += 256) Ts[i] = T[i];
  if (tid < C_ * 4) Ds[tid] = dims[tid];
  __syncthreads();

  int gid = blockIdx.x * 256 + tid;
  int n = gid >> 6;
  int c = gid & 63;

  const float* row = I + (size_t)n * D_;
  float x[4];
  #pragma unroll
  for (int j = 0; j < 4; ++j) x[j] = row[Ds[c * 4 + j]];

  const int lvl[NODES_] = {0,1,1,2,2,2,2,3,3,3,3,3,3,3,3};
  float s[NODES_];
  #pragma unroll
  for (int i = 0; i < NODES_; ++i) {
    float h = x[lvl[i]] - Ts[c * NODES_ + i];
    s[i] = (h > 0.f) ? 1.f : ((h < 0.f) ? -1.f : 0.f);
  }

  float invt = 1.0f / temp[0];
  float best = -3.4e38f;
  int bk = 0;
  #pragma unroll
  for (int k = 0; k < K_; ++k) {
    float b = 0.f;
    #pragma unroll
    for (int l = 0; l < 4; ++l) {
      int node = (1 << l) - 1 + (k >> (4 - l));
      int bit  = (k >> (3 - l)) & 1;
      b += bit ? s[node] : -s[node];
    }
    float bt = b * invt;
    if (bt > best) { best = bt; bk = k; }   // strict >: first-max, like jnp.argmax
  }
  ksel[(size_t)n * C_ + c] = (unsigned char)bk;
}

// ---------------------------------------------------------------------------
// Kernel 2: pre-swizzle L (fp32 [M][C][K]) into WMMA B-fragment layout as
// f16 hi + f16 residual lo. Fragment half index:
//   idx = ((mt*32 + chunk)*32 + lane)*16 + e,  K = (lane>=16?8:0)+e+(e>=8?8:0),
//   c = 2*chunk + (K>>4), k = K&15, m = mt*16 + (lane&15)
// ---------------------------------------------------------------------------
__global__ __launch_bounds__(256) void halut_lprep(
    const float* __restrict__ L,
    _Float16* __restrict__ bhi, _Float16* __restrict__ blo)
{
  int idx   = blockIdx.x * 256 + threadIdx.x;   // 0..131071
  int e     = idx & 15;
  int lane  = (idx >> 4) & 31;
  int chunk = (idx >> 9) & 31;
  int mt    = idx >> 14;
  int Kg    = ((lane >= 16) ? 8 : 0) + e + ((e >= 8) ? 8 : 0);
  int c     = chunk * 2 + (Kg >> 4);
  int k     = Kg & 15;
  int m     = mt * 16 + (lane & 15);
  float v   = L[(size_t)m * CK_ + c * K_ + k];
  _Float16 h = (_Float16)v;
  bhi[idx] = h;
  blo[idx] = (_Float16)(v - (float)h);
}

// ---------------------------------------------------------------------------
// Kernel 3: out[N][M] = E @ L^T via v_wmma_f32_16x16x32_f16, hi/lo split.
// Block = 16 n-rows, 8 waves; wave w owns m-tile w (16x16 output tile).
// A fragment built on the fly from ksel bytes staged in LDS (one-hot in f16).
// ---------------------------------------------------------------------------
__global__ __launch_bounds__(256) void halut_gemm(
    const unsigned char* __restrict__ ksel,
    const v16h* __restrict__ bhi, const v16h* __restrict__ blo,
    float* __restrict__ out)
{
  __shared__ unsigned char ks[16 * C_];   // 1024 B: ksel rows for this n-tile
  int tid = threadIdx.x;
  int n_base = blockIdx.x * 16;
  ((uint32_t*)ks)[tid] = ((const uint32_t*)(ksel + (size_t)n_base * C_))[tid];
  __syncthreads();

  int wave = tid >> 5;            // m-tile
  int lane = tid & 31;
  int r    = lane & 15;           // A row / C-D column within tile
  int hs   = (lane & 16) ? 8 : 0; // half-select for A K-map and C/D M-map

  const v16h* bh = bhi + (size_t)wave * NCHUNK_ * 32 + lane;
  const v16h* bl = blo + (size_t)wave * NCHUNK_ * 32 + lane;
  const unsigned char* kr = ks + r * C_;

  v8f acc = {};
  #pragma unroll 4
  for (int chunk = 0; chunk < NCHUNK_; ++chunk) {
    int k0 = kr[2 * chunk];       // selected leaf for c = 2*chunk
    int k1 = kr[2 * chunk + 1];   // selected leaf for c = 2*chunk + 1
    v16h a;
    #pragma unroll
    for (int e = 0; e < 8; ++e)       // K = hs + e        (c0 block)
      a[e] = ((hs + e) == k0) ? (_Float16)1.0f : (_Float16)0.0f;
    #pragma unroll
    for (int e = 8; e < 16; ++e)      // K = 16 + hs + e-8 (c1 block)
      a[e] = ((hs + e - 8) == k1) ? (_Float16)1.0f : (_Float16)0.0f;

    v16h Bh = bh[chunk * 32];
    v16h Bl = bl[chunk * 32];
    acc = __builtin_amdgcn_wmma_f32_16x16x32_f16(false, a, false, Bh,
                                                 (short)0, acc, false, false);
    acc = __builtin_amdgcn_wmma_f32_16x16x32_f16(false, a, false, Bl,
                                                 (short)0, acc, false, false);
  }

  // C/D layout: VGPR v -> M = hs + v, N = lane%16
  float* o = out + (size_t)(n_base + hs) * M_ + wave * 16 + r;
  #pragma unroll
  for (int v = 0; v < 8; ++v)
    o[(size_t)v * M_] = acc[v];
}

// ---------------------------------------------------------------------------
extern "C" void kernel_launch(void* const* d_in, const int* in_sizes, int n_in,
                              void* d_out, int out_size, void* d_ws, size_t ws_size,
                              hipStream_t stream) {
  (void)in_sizes; (void)n_in; (void)out_size; (void)ws_size;
  const float* I    = (const float*)d_in[0];
  const float* T    = (const float*)d_in[1];
  const float* L    = (const float*)d_in[2];
  // d_in[3] = S, d_in[4] = B : block-diagonal selection matrices, folded into
  // the tree-descent logic (entries are exactly 0/±1), not needed at runtime.
  const int*   dims = (const int*)d_in[5];
  const float* temp = (const float*)d_in[6];
  float* out = (float*)d_out;

  unsigned char* ksel = (unsigned char*)d_ws;                       // 2 MB
  _Float16* bhi = (_Float16*)((char*)d_ws + (size_t)N_ * C_);       // 256 KB
  _Float16* blo = bhi + BFRAG_HALVES;                               // 256 KB

  halut_ksel <<<(N_ * C_) / 256, 256, 0, stream>>>(I, T, dims, temp, ksel);
  halut_lprep<<<BFRAG_HALVES / 256, 256, 0, stream>>>(L, bhi, blo);
  halut_gemm <<<N_ / 16, 256, 0, stream>>>(ksel, (const v16h*)bhi,
                                           (const v16h*)blo, out);
}
